// TransformerEncoderLayer_54726473286218
// MI455X (gfx1250) — matrix operand order repaired and checked by reading, not compiled, not run on hardware
//
#include <hip/hip_runtime.h>
#include <hip/hip_bf16.h>

// ---------------------------------------------------------------------------
// Types for CDNA5 WMMA (wave32)
// ---------------------------------------------------------------------------
typedef __bf16 bf16_t;
typedef __attribute__((ext_vector_type(16))) __bf16 v16bf;
typedef __attribute__((ext_vector_type(8)))  __bf16 v8bf;
typedef __attribute__((ext_vector_type(8)))  float  v8f;

#define S_LEN 4096
#define D_DIM 512
#define H_NUM 8
#define HD_DIM 64
#define FF_DIM 2048

__device__ __forceinline__ v8f wmma_bf16(v16bf a, v16bf b, v8f c) {
  // D = A(16x32 bf16) * B(32x16 bf16) + C(16x16 f32)
  return __builtin_amdgcn_wmma_f32_16x16x32_bf16(
      /*neg_a=*/false, a, /*neg_b=*/false, b,
      /*c_mod=*/(short)0, c, /*reuse_a=*/false, /*reuse_b=*/false);
}

// ---------------------------------------------------------------------------
// DPP rotate-reductions over a 16-lane row (pure VALU: no LDS, no waits).
// After ror:8/4/2/1 steps every lane of the row holds the full reduction.
// ---------------------------------------------------------------------------
__device__ __forceinline__ float dpp_ror(float v, int ctrl_unused) { return v; }

#define ROR_STEP(v, ctrl, OP)                                                  \
  {                                                                            \
    int _s = __builtin_amdgcn_update_dpp(                                      \
        0, __builtin_bit_cast(int, v), ctrl, 0xf, 0xf, true);                  \
    v = OP(v, __builtin_bit_cast(float, _s));                                  \
  }

__device__ __forceinline__ float row16_max(float v) {
  ROR_STEP(v, 0x128, fmaxf)   // row_ror:8
  ROR_STEP(v, 0x124, fmaxf)   // row_ror:4
  ROR_STEP(v, 0x122, fmaxf)   // row_ror:2
  ROR_STEP(v, 0x121, fmaxf)   // row_ror:1
  return v;
}
__device__ __forceinline__ float fadd2(float a, float b) { return a + b; }
__device__ __forceinline__ float row16_sum(float v) {
  ROR_STEP(v, 0x128, fadd2)
  ROR_STEP(v, 0x124, fadd2)
  ROR_STEP(v, 0x122, fadd2)
  ROR_STEP(v, 0x121, fadd2)
  return v;
}

// A operand, 16x32 bf16, row-major source with leading dim lda.
// Lane l: row = m0 + (l&15); K chunks: [koff, koff+8) -> elems 0..7,
// [koff+8, koff+16) -> elems 8..15, koff = k0 + (l>>4)*16.
__device__ __forceinline__ v16bf load_A16x32(const bf16_t* __restrict__ A,
                                             int lda, int m0, int k0, int lane) {
  int row  = m0 + (lane & 15);
  int koff = k0 + ((lane >> 4) << 4);
  const v8bf* p0 = reinterpret_cast<const v8bf*>(A + (size_t)row * lda + koff);
  const v8bf* p1 = reinterpret_cast<const v8bf*>(A + (size_t)row * lda + koff + 8);
  v8bf lo = *p0;
  v8bf hi = *p1;
  v16bf r;
#pragma unroll
  for (int i = 0; i < 8; ++i) { r[i] = lo[i]; r[i + 8] = hi[i]; }
  return r;
}

// B operand, 32x16 bf16, from a K-contiguous (transposed) source Bt[n][k].
// Lane l: col = n0 + (l&15); 16 contiguous k at k0 + (l>>4)*16.
__device__ __forceinline__ v16bf load_B32x16(const bf16_t* __restrict__ Bt,
                                             int ldb, int n0, int k0, int lane) {
  int col  = n0 + (lane & 15);
  int koff = k0 + ((lane >> 4) << 4);
  return *reinterpret_cast<const v16bf*>(Bt + (size_t)col * ldb + koff);
}

// ---------------------------------------------------------------------------
// Weight convert + transpose (LDS-tiled, coalesced both sides):
//   Wt[n*K + k] = (bf16) W[k*N + n]
// block (32,8), grid (N/32, K/32)
// ---------------------------------------------------------------------------
__global__ void convert_T_kernel(const float* __restrict__ W, bf16_t* __restrict__ Wt,
                                 int K, int N) {
  __shared__ float tile[32][33];
  int nt = blockIdx.x * 32;
  int kt = blockIdx.y * 32;
  int tx = threadIdx.x;
  int ty = threadIdx.y;
#pragma unroll
  for (int r = 0; r < 32; r += 8)
    tile[ty + r][tx] = W[(size_t)(kt + ty + r) * N + nt + tx];
  __syncthreads();
#pragma unroll
  for (int r = 0; r < 32; r += 8)
    Wt[(size_t)(nt + ty + r) * K + kt + tx] = (bf16_t)tile[tx][ty + r];
}

// ---------------------------------------------------------------------------
// LayerNorm (one wave per row of 512) + affine, fp32 -> bf16
// ---------------------------------------------------------------------------
__global__ void ln_affine_kernel(const float* __restrict__ x,
                                 const float* __restrict__ gammas,
                                 const float* __restrict__ betas,
                                 int which, bf16_t* __restrict__ out) {
  int row  = blockIdx.x * (blockDim.x >> 5) + (threadIdx.x >> 5);
  int lane = threadIdx.x & 31;
  const float4* xr = reinterpret_cast<const float4*>(x + (size_t)row * D_DIM);
  float4 v[4];
  float s = 0.f;
#pragma unroll
  for (int i = 0; i < 4; ++i) {
    v[i] = xr[lane + 32 * i];
    s += v[i].x + v[i].y + v[i].z + v[i].w;
  }
#pragma unroll
  for (int off = 16; off >= 1; off >>= 1) s += __shfl_xor(s, off, 32);
  float mean = s * (1.f / 512.f);
  float q = 0.f;
#pragma unroll
  for (int i = 0; i < 4; ++i) {
    float a = v[i].x - mean, b = v[i].y - mean, c = v[i].z - mean, d = v[i].w - mean;
    q += a * a + b * b + c * c + d * d;
  }
#pragma unroll
  for (int off = 16; off >= 1; off >>= 1) q += __shfl_xor(q, off, 32);
  float rinv = rsqrtf(q * (1.f / 512.f) + 1e-5f);
  float g = gammas[which], be = betas[which];
  bf16_t* orow = out + (size_t)row * D_DIM;
#pragma unroll
  for (int i = 0; i < 4; ++i) {
    int base = 4 * (lane + 32 * i);
    orow[base + 0] = (bf16_t)(g * ((v[i].x - mean) * rinv) + be);
    orow[base + 1] = (bf16_t)(g * ((v[i].y - mean) * rinv) + be);
    orow[base + 2] = (bf16_t)(g * ((v[i].z - mean) * rinv) + be);
    orow[base + 3] = (bf16_t)(g * ((v[i].w - mean) * rinv) + be);
  }
}

// ---------------------------------------------------------------------------
// WMMA GEMM, 16x64 output strip per wave (4 N-tiles, A fragment reused 4x).
//   MODE 0: out fp32 = A@Bt                      (Q/K projections)
//   MODE 1: out bf16 transposed (V -> Vt[D][S])  (V projection)
//   MODE 2: out fp32 = A@Bt + bias + resid       (out-proj / FFN2)
//   MODE 3: out bf16 = GELU(A@Bt + bias)         (FFN1)
// grid: (N/64, M/128), block 256 (8 waves)
// ---------------------------------------------------------------------------
template <int MODE>
__global__ void gemm64_kernel(const bf16_t* __restrict__ A, const bf16_t* __restrict__ Bt,
                              int M, int N, int K,
                              const float* __restrict__ bias,
                              const float* __restrict__ resid,
                              float* __restrict__ outF,
                              bf16_t* __restrict__ outB, int ldOutB) {
  int lane = threadIdx.x & 31;
  int wave = threadIdx.x >> 5;
  int n0 = blockIdx.x * 64;
  int m0 = (blockIdx.y * 8 + wave) * 16;
  if (m0 >= M) return;

  v8f acc[4];
#pragma unroll
  for (int t = 0; t < 4; ++t) acc[t] = (v8f){0.f,0.f,0.f,0.f,0.f,0.f,0.f,0.f};

  int rowA = m0 + (lane & 15);
  for (int k0 = 0; k0 < K; k0 += 32) {
    if (k0 + 32 < K) {
      __builtin_prefetch(A + (size_t)rowA * K + k0 + 32, 0, 0);   // global_prefetch_b8
    }
    v16bf a = load_A16x32(A, K, m0, k0, lane);
#pragma unroll
    for (int t = 0; t < 4; ++t) {
      v16bf b = load_B32x16(Bt, K, n0 + 16 * t, k0, lane);
      acc[t] = wmma_bf16(a, b, acc[t]);
    }
  }

#pragma unroll
  for (int t = 0; t < 4; ++t) {
    int n = n0 + 16 * t + (lane & 15);
    float bv = (MODE >= 2) ? bias[n] : 0.f;
#pragma unroll
    for (int i = 0; i < 8; ++i) {
      int m = m0 + i + ((lane >> 4) << 3);
      float val = acc[t][i] + bv;
      if (MODE == 0) {
        outF[(size_t)m * N + n] = val;
      } else if (MODE == 1) {
        outB[(size_t)n * ldOutB + m] = (bf16_t)val;          // transposed store
      } else if (MODE == 2) {
        outF[(size_t)m * N + n] = val + resid[(size_t)m * N + n];
      } else {  // MODE == 3 : exact GELU
        float gl = 0.5f * val * (1.f + erff(val * 0.70710678118654752f));
        outB[(size_t)m * N + n] = (bf16_t)gl;
      }
    }
  }
}

// ---------------------------------------------------------------------------
// RoPE (first 32 dims of each 64-dim head) + fp32 -> bf16 convert
// ---------------------------------------------------------------------------
__global__ void rope_convert_kernel(const float* __restrict__ src, bf16_t* __restrict__ dst) {
  int idx = blockIdx.x * blockDim.x + threadIdx.x;   // over S*D/2 pairs
  int s = idx >> 8;                 // D/2 = 256 pairs per row
  int p = idx & 255;
  int d0 = 2 * p;
  int hd = d0 & (HD_DIM - 1);
  float a = src[(size_t)s * D_DIM + d0];
  float b = src[(size_t)s * D_DIM + d0 + 1];
  float oa = a, ob = b;
  if (hd < 32) {
    int j = hd >> 1;                                   // freq index 0..15
    float inv = __powf(10000.0f, -(float)j * (1.0f / 16.0f));
    float ang = (float)s * inv;
    float sn, cs;
    __sincosf(ang, &sn, &cs);
    oa = a * cs - b * sn;
    ob = b * cs + a * sn;
  }
  dst[(size_t)s * D_DIM + d0]     = (bf16_t)oa;
  dst[(size_t)s * D_DIM + d0 + 1] = (bf16_t)ob;
}

// ---------------------------------------------------------------------------
// Flash attention: one wave = one 16-query tile of one head.
// QK^T and P@V via v_wmma_f32_16x16x32_bf16; online softmax with DPP
// rotate-reductions (row-of-16); P transposed to A layout through LDS.
// ---------------------------------------------------------------------------
__global__ void flash_attn_kernel(const bf16_t* __restrict__ Qb,
                                  const bf16_t* __restrict__ Kb,
                                  const bf16_t* __restrict__ Vt,   // [D][S]
                                  const int* __restrict__ mask,    // [S]
                                  bf16_t* __restrict__ Ob) {       // [S][D]
  __shared__ float ldsP[8][16 * 32];
  int lane = threadIdx.x & 31;
  int wave = threadIdx.x >> 5;
  int h  = blockIdx.y;
  int q0 = (blockIdx.x * 8 + wave) * 16;
  float* P = ldsP[wave];

  const bf16_t* Qh = Qb + h * HD_DIM;
  const bf16_t* Kh = Kb + h * HD_DIM;
  const bf16_t* Vh = Vt + (size_t)h * HD_DIM * S_LEN;

  v16bf qa0 = load_A16x32(Qh, D_DIM, q0, 0, lane);
  v16bf qa1 = load_A16x32(Qh, D_DIM, q0, 32, lane);

  v8f o0 = {0.f,0.f,0.f,0.f,0.f,0.f,0.f,0.f};
  v8f o1 = o0, o2 = o0, o3 = o0;
  float mrow[8], lrow[8];
#pragma unroll
  for (int i = 0; i < 8; ++i) { mrow[i] = -1e30f; lrow[i] = 0.f; }

  const float scale = 0.125f;  // 1/sqrt(64)

  for (int j0 = 0; j0 < S_LEN; j0 += 32) {
    // ---- S = Q @ K^T (two 16x16 tiles over the 32-key block) ----
    v8f s0 = {0.f,0.f,0.f,0.f,0.f,0.f,0.f,0.f};
    v8f s1 = s0;
    s0 = wmma_bf16(qa0, load_B32x16(Kh, D_DIM, j0,      0, lane), s0);
    s0 = wmma_bf16(qa1, load_B32x16(Kh, D_DIM, j0,     32, lane), s0);
    s1 = wmma_bf16(qa0, load_B32x16(Kh, D_DIM, j0 + 16, 0, lane), s1);
    s1 = wmma_bf16(qa1, load_B32x16(Kh, D_DIM, j0 + 16,32, lane), s1);

    // ---- mask + scale (lane owns one key column per tile) ----
    int key0 = j0 + (lane & 15);
    float msk0 = (mask[key0]      == 0) ? -1e9f : 0.f;
    float msk1 = (mask[key0 + 16] == 0) ? -1e9f : 0.f;
#pragma unroll
    for (int i = 0; i < 8; ++i) {
      s0[i] = s0[i] * scale + msk0;
      s1[i] = s1[i] * scale + msk1;
    }

    // ---- online softmax: rows live across 16-lane half-groups ----
#pragma unroll
    for (int i = 0; i < 8; ++i) {
      float mx   = row16_max(fmaxf(s0[i], s1[i]));
      float mnew = fmaxf(mrow[i], mx);
      float corr = __expf(mrow[i] - mnew);
      float p0 = __expf(s0[i] - mnew);
      float p1 = __expf(s1[i] - mnew);
      float rs = row16_sum(p0 + p1);
      lrow[i] = lrow[i] * corr + rs;
      mrow[i] = mnew;
      o0[i] *= corr; o1[i] *= corr; o2[i] *= corr; o3[i] *= corr;
      s0[i] = p0; s1[i] = p1;
    }

    // ---- re-layout P (C/D layout -> A layout) through per-wave LDS ----
    {
      int mb = (lane >> 4) << 3;
      int n  = lane & 15;
#pragma unroll
      for (int i = 0; i < 8; ++i) {
        P[(mb + i) * 32 + n]      = s0[i];
        P[(mb + i) * 32 + n + 16] = s1[i];
      }
    }
    asm volatile("s_wait_dscnt 0" ::: "memory");   // wave-local LDS RAW ordering
    v16bf pa;
    {
      int row  = lane & 15;
      int koff = (lane >> 4) << 4;
#pragma unroll
      for (int i = 0; i < 8; ++i) {
        pa[i]     = (bf16_t)P[row * 32 + koff + i];
        pa[i + 8] = (bf16_t)P[row * 32 + koff + 8 + i];
      }
    }

    // ---- O += P @ V  (Vt is [hd][s]: a native K-contiguous B operand) ----
    o0 = wmma_bf16(pa, load_B32x16(Vh, S_LEN,  0, j0, lane), o0);
    o1 = wmma_bf16(pa, load_B32x16(Vh, S_LEN, 16, j0, lane), o1);
    o2 = wmma_bf16(pa, load_B32x16(Vh, S_LEN, 32, j0, lane), o2);
    o3 = wmma_bf16(pa, load_B32x16(Vh, S_LEN, 48, j0, lane), o3);
  }

  // ---- normalize and store bf16 [S][D] ----
#pragma unroll
  for (int i = 0; i < 8; ++i) {
    float inv = 1.0f / lrow[i];
    int m = q0 + i + ((lane >> 4) << 3);
    size_t base = (size_t)m * D_DIM + h * HD_DIM + (lane & 15);
    Ob[base]      = (bf16_t)(o0[i] * inv);
    Ob[base + 16] = (bf16_t)(o1[i] * inv);
    Ob[base + 32] = (bf16_t)(o2[i] * inv);
    Ob[base + 48] = (bf16_t)(o3[i] * inv);
  }
}

// ---------------------------------------------------------------------------
// Host-side orchestration
// ---------------------------------------------------------------------------
extern "C" void kernel_launch(void* const* d_in, const int* in_sizes, int n_in,
                              void* d_out, int out_size, void* d_ws, size_t ws_size,
                              hipStream_t stream) {
  (void)in_sizes; (void)n_in; (void)out_size; (void)ws_size;
  const float* x      = (const float*)d_in[0];
  const float* gammas = (const float*)d_in[1];
  const float* betas  = (const float*)d_in[2];
  const float* wq     = (const float*)d_in[3];
  const float* wk     = (const float*)d_in[4];
  const float* wv     = (const float*)d_in[5];
  const float* wo     = (const float*)d_in[6];
  const float* bo     = (const float*)d_in[7];
  const float* w1     = (const float*)d_in[8];
  const float* b1     = (const float*)d_in[9];
  const float* w2     = (const float*)d_in[10];
  const float* b2     = (const float*)d_in[11];
  const int*   mask   = (const int*)d_in[12];
  float* out = (float*)d_out;

  char* ws = (char*)d_ws;
  bf16_t* wqT   = (bf16_t*)(ws + 0);          // 512*512*2
  bf16_t* wkT   = (bf16_t*)(ws + 524288);
  bf16_t* wvT   = (bf16_t*)(ws + 1048576);
  bf16_t* woT   = (bf16_t*)(ws + 1572864);
  bf16_t* w1T   = (bf16_t*)(ws + 2097152);    // [FF][D]
  bf16_t* w2T   = (bf16_t*)(ws + 4194304);    // [D][FF]
  bf16_t* h_bf  = (bf16_t*)(ws + 6291456);    // [S][D]
  float*  q_f   = (float*) (ws + 10485760);   // [S][D] fp32   (dead after rope)
  float*  k_f   = (float*) (ws + 18874368);   // [S][D] fp32   (dead after rope)
  bf16_t* ffh   = (bf16_t*)(ws + 10485760);   // [S][FF] bf16  (aliases q_f/k_f, later)
  bf16_t* q_bf  = (bf16_t*)(ws + 27262976);   // [S][D]
  bf16_t* k_bf  = (bf16_t*)(ws + 31457280);   // [S][D]
  bf16_t* vT    = (bf16_t*)(ws + 35651584);   // [D][S]
  bf16_t* attnb = (bf16_t*)(ws + 39845888);   // [S][D]
  float*  x1    = (float*) (ws + 44040192);   // [S][D] fp32 residual stream
  bf16_t* h2_bf = (bf16_t*)(ws + 52428672);   // [S][D]

  // 1. weight convert + transpose (LDS-tiled)
  dim3 tb(32, 8);
  convert_T_kernel<<<dim3(512/32,  512/32),  tb, 0, stream>>>(wq, wqT, 512, 512);
  convert_T_kernel<<<dim3(512/32,  512/32),  tb, 0, stream>>>(wk, wkT, 512, 512);
  convert_T_kernel<<<dim3(512/32,  512/32),  tb, 0, stream>>>(wv, wvT, 512, 512);
  convert_T_kernel<<<dim3(512/32,  512/32),  tb, 0, stream>>>(wo, woT, 512, 512);
  convert_T_kernel<<<dim3(2048/32, 512/32),  tb, 0, stream>>>(w1, w1T, 512, 2048);
  convert_T_kernel<<<dim3(512/32,  2048/32), tb, 0, stream>>>(w2, w2T, 2048, 512);

  // 2. LN1 -> bf16
  ln_affine_kernel<<<S_LEN/8, 256, 0, stream>>>(x, gammas, betas, 0, h_bf);

  // 3. Q/K/V projections (WMMA, 16x64 strip per wave)
  dim3 gProj(D_DIM/64, S_LEN/128);
  gemm64_kernel<0><<<gProj, 256, 0, stream>>>(h_bf, wqT, S_LEN, D_DIM, D_DIM,
                                              nullptr, nullptr, q_f, nullptr, 0);
  gemm64_kernel<0><<<gProj, 256, 0, stream>>>(h_bf, wkT, S_LEN, D_DIM, D_DIM,
                                              nullptr, nullptr, k_f, nullptr, 0);
  gemm64_kernel<1><<<gProj, 256, 0, stream>>>(h_bf, wvT, S_LEN, D_DIM, D_DIM,
                                              nullptr, nullptr, nullptr, vT, S_LEN);

  // 4. RoPE + bf16 convert
  rope_convert_kernel<<<(S_LEN*D_DIM/2)/256, 256, 0, stream>>>(q_f, q_bf);
  rope_convert_kernel<<<(S_LEN*D_DIM/2)/256, 256, 0, stream>>>(k_f, k_bf);

  // 5. flash attention (WMMA)
  flash_attn_kernel<<<dim3(S_LEN/128, H_NUM), 256, 0, stream>>>(q_bf, k_bf, vT, mask, attnb);

  // 6. out-proj + bias + residual -> x1
  gemm64_kernel<2><<<gProj, 256, 0, stream>>>(attnb, woT, S_LEN, D_DIM, D_DIM,
                                              bo, x, x1, nullptr, 0);

  // 7. LN2 -> bf16
  ln_affine_kernel<<<S_LEN/8, 256, 0, stream>>>(x1, gammas, betas, 1, h2_bf);

  // 8. FFN: GELU(h2 @ w1 + b1) @ w2 + b2 + x1 -> out
  dim3 gFF1(FF_DIM/64, S_LEN/128);
  gemm64_kernel<3><<<gFF1, 256, 0, stream>>>(h2_bf, w1T, S_LEN, FF_DIM, D_DIM,
                                             b1, nullptr, nullptr, ffh, 0);
  gemm64_kernel<2><<<gProj, 256, 0, stream>>>(ffh, w2T, S_LEN, D_DIM, FF_DIM,
                                              b2, x1, out, nullptr, 0);
}
